// Indexer_12421045420394
// MI455X (gfx1250) — matrix operand order repaired and checked by reading, not compiled, not run on hardware
//
#include <hip/hip_runtime.h>
#include <hip/hip_bf16.h>

// ---------------------------------------------------------------------------
// DeepSeek-style indexer prologue, B=2 S=4096 D=7168 LORA=1536 H=64 DH=128.
// Outputs concatenated in d_out (float32):
//   q       : 67108864   (offset 0)
//   weights :   524288   (offset 67108864)
//   k       :  1048576   (offset 67633152)
//   end_pos :        1   (offset 68681728)
// ---------------------------------------------------------------------------

typedef __bf16 bf16_t;
typedef bf16_t v16bf __attribute__((ext_vector_type(16)));
typedef bf16_t v8bf  __attribute__((ext_vector_type(8)));
typedef float  v8f   __attribute__((ext_vector_type(8)));

static constexpr int kS    = 4096;
static constexpr int kD    = 7168;
static constexpr int kLora = 1536;
static constexpr int kH    = 64;
static constexpr int kDH   = 128;

static constexpr int LDK = 48;   // ushort stride of K-tiles (KTILE=32, padded, 16B multiple)
static constexpr int LDB = 128;  // ushort stride of 128-wide bf16 tiles

#define WMMA_BF16(Afrag, Bfrag, Cacc) \
  __builtin_amdgcn_wmma_f32_16x16x32_bf16(false, (Afrag), false, (Bfrag), (short)0, (Cacc), false, false)

static __device__ __forceinline__ unsigned short f2bf(float f) {
  union { float f; unsigned int u; } v; v.f = f;
  unsigned int u = v.u;
  u += 0x7fffu + ((u >> 16) & 1u);      // round-to-nearest-even
  return (unsigned short)(u >> 16);
}

// A fragment (16x32 bf16, row-major [M][K] in LDS).
static __device__ __forceinline__ v16bf frag_a(const unsigned short* base, int ldk, int lane) {
  const unsigned short* p = base + (lane & 15) * ldk + ((lane >> 4) << 3);
  v8bf lo = *(const v8bf*)p;
  v8bf hi = *(const v8bf*)(p + 16);
  return __builtin_shufflevector(lo, hi, 0,1,2,3,4,5,6,7,8,9,10,11,12,13,14,15);
}

// B fragment (32x16 bf16, stored [N][K] row-major in LDS).
static __device__ __forceinline__ v16bf frag_b(const unsigned short* base, int ldk, int lane) {
  const unsigned short* p = base + (lane & 15) * ldk + ((lane >> 4) << 3);
  v8bf lo = *(const v8bf*)p;
  v8bf hi = *(const v8bf*)(p + 16);
  return __builtin_shufflevector(lo, hi, 0,1,2,3,4,5,6,7,8,9,10,11,12,13,14,15);
}

static __device__ __forceinline__ void fill_hadamard(unsigned short* Hbf, int tid, int nthreads) {
  // Sylvester Hadamard, H[r][c] = (-1)^popc(r&c) / sqrt(128); symmetric.
  for (int i = tid; i < 128 * 128; i += nthreads) {
    int r = i >> 7, c = i & 127;
    float v = (__popc(r & c) & 1) ? -0.08838834764831845f : 0.08838834764831845f;
    Hbf[r * LDB + c] = f2bf(v);
  }
}

// ---------------------------------------------------------------------------
// Kernel 1: q = RoPE(qr @ wq_b_w^T) @ Hadamard, fused per (M-tile, head).
// grid = 64 mtiles * 64 heads; block = 256 (8 waves). LDS double-buffered.
// ---------------------------------------------------------------------------
__global__ __launch_bounds__(256) void q_fused_kernel(
    const float* __restrict__ qr, const float* __restrict__ wq,
    const float* __restrict__ fc, const float* __restrict__ fs,
    float* __restrict__ qout)
{
  extern __shared__ char smem[];
  unsigned short* As  = (unsigned short*)(smem);            // 2 x 128*48*2 = 24576
  unsigned short* Bs  = (unsigned short*)(smem + 24576);    // 2 x 128*48*2 = 24576
  unsigned short* qbf = (unsigned short*)(smem + 49152);    // 128*128*2 = 32768
  unsigned short* Hbf = (unsigned short*)(smem + 81920);    // 32768
  constexpr int TSZ = 128 * LDK;                            // ushorts per buffer

  const int tid  = threadIdx.x;
  const int lane = tid & 31;
  const int wave = tid >> 5;
  const int m0   = (blockIdx.x & 63) * 128;
  const int h    = blockIdx.x >> 6;
  const int n0   = h * 128;

  fill_hadamard(Hbf, tid, 256);

  v8f acc[8];
  for (int nt = 0; nt < 8; ++nt)
    for (int e = 0; e < 8; ++e) acc[nt][e] = 0.0f;

  float4 ra[4], rb[4];
  auto load_regs = [&](int k0) {
    for (int j = 0; j < 4; ++j) {
      int idx = tid + (j << 8);              // 0..1023
      int r   = idx >> 3;                    // 0..127
      int c4  = (idx & 7) << 2;              // 0..28
      const float* pa = qr + (size_t)(m0 + r) * kLora + k0 + c4;
      const float* pb = wq + (size_t)(n0 + r) * kLora + k0 + c4;
      __builtin_prefetch(pa + 32, 0, 3);
      __builtin_prefetch(pb + 32, 0, 3);
      ra[j] = *(const float4*)pa;
      rb[j] = *(const float4*)pb;
    }
  };
  auto store_lds = [&](int buf) {
    unsigned short* A  = As + buf * TSZ;
    unsigned short* Bt = Bs + buf * TSZ;
    for (int j = 0; j < 4; ++j) {
      int idx = tid + (j << 8);
      int r   = idx >> 3;
      int c4  = (idx & 7) << 2;
      unsigned short* sa = A + r * LDK + c4;
      sa[0] = f2bf(ra[j].x); sa[1] = f2bf(ra[j].y); sa[2] = f2bf(ra[j].z); sa[3] = f2bf(ra[j].w);
      unsigned short* sb = Bt + r * LDK + c4;
      sb[0] = f2bf(rb[j].x); sb[1] = f2bf(rb[j].y); sb[2] = f2bf(rb[j].z); sb[3] = f2bf(rb[j].w);
    }
  };

  constexpr int NK = kLora / 32;             // 48
  load_regs(0);
  store_lds(0);
  __syncthreads();
  for (int kt = 0; kt < NK; ++kt) {
    const int cur = kt & 1;
    const bool more = (kt + 1) < NK;
    if (more) load_regs((kt + 1) * 32);      // global loads overlap with WMMA below
    v16bf a = frag_a(As + cur * TSZ + wave * 16 * LDK, LDK, lane);
    v16bf bf[8];
    for (int nt = 0; nt < 8; ++nt)
      bf[nt] = frag_b(Bs + cur * TSZ + nt * 16 * LDK, LDK, lane);
    for (int nt = 0; nt < 8; ++nt)
      acc[nt] = WMMA_BF16(a, bf[nt], acc[nt]);
    if (more) store_lds(cur ^ 1);
    __syncthreads();
  }

  // ---- RoPE in registers (pairs (d, d+32) share lane & slot j) + bf16 to LDS ----
  {
    const int c0 = lane & 15;
    for (int j = 0; j < 8; ++j) {
      const int r = wave * 16 + j + ((lane >> 4) << 3);
      const int s = (m0 + r) & (kS - 1);
      float c0v = fc[s * 32 + c0],      s0v = fs[s * 32 + c0];
      float c1v = fc[s * 32 + 16 + c0], s1v = fs[s * 32 + 16 + c0];
      float n0 = acc[0][j] * c0v - acc[2][j] * s0v;
      float n2 = acc[0][j] * s0v + acc[2][j] * c0v;
      float n1 = acc[1][j] * c1v - acc[3][j] * s1v;
      float n3 = acc[1][j] * s1v + acc[3][j] * c1v;
      unsigned short* q = qbf + r * LDB + c0;
      q[0]   = f2bf(n0);         q[16]  = f2bf(n1);
      q[32]  = f2bf(n2);         q[48]  = f2bf(n3);
      q[64]  = f2bf(acc[4][j]);  q[80]  = f2bf(acc[5][j]);
      q[96]  = f2bf(acc[6][j]);  q[112] = f2bf(acc[7][j]);
    }
  }
  __syncthreads();

  // ---- Hadamard via WMMA: (16x128 strip) @ (128x128) ----
  v8f hacc[8];
  for (int nt = 0; nt < 8; ++nt)
    for (int e = 0; e < 8; ++e) hacc[nt][e] = 0.0f;
  for (int kk = 0; kk < 4; ++kk) {
    v16bf a = frag_a(qbf + wave * 16 * LDB + kk * 32, LDB, lane);
    v16bf bf[8];
    for (int nt = 0; nt < 8; ++nt)
      bf[nt] = frag_b(Hbf + nt * 16 * LDB + kk * 32, LDB, lane);
    for (int nt = 0; nt < 8; ++nt)
      hacc[nt] = WMMA_BF16(a, bf[nt], hacc[nt]);
  }
  for (int nt = 0; nt < 8; ++nt)
    for (int j = 0; j < 8; ++j) {
      int m = m0 + wave * 16 + j + ((lane >> 4) << 3);
      qout[(size_t)m * (kH * kDH) + h * kDH + nt * 16 + (lane & 15)] = hacc[nt][j];
    }
}

// ---------------------------------------------------------------------------
// Kernel 2: fused x-projections, N = 192 = 128 (k) + 64 (weights); x read once.
// k: layernorm (shfl-reduce, wave-local rows) + RoPE in registers + Hadamard-WMMA.
// M-tile = 64 rows, block = 128 (4 waves) -> 128 WGs for HBM-side parallelism.
// ---------------------------------------------------------------------------
__global__ __launch_bounds__(128) void kw_fused_kernel(
    const float* __restrict__ x,  const float* __restrict__ wk,
    const float* __restrict__ wp, const float* __restrict__ g,
    const float* __restrict__ bb, const float* __restrict__ fc,
    const float* __restrict__ fs,
    float* __restrict__ wout, float* __restrict__ kout)
{
  extern __shared__ char smem[];
  unsigned short* As  = (unsigned short*)(smem);            // 2 x  64*48*2 = 12288
  unsigned short* Bs  = (unsigned short*)(smem + 12288);    // 2 x 192*48*2 = 36864
  unsigned short* kbf = (unsigned short*)(smem + 49152);    //  64*128*2 = 16384
  unsigned short* Hbf = (unsigned short*)(smem + 65536);    // 32768; total 98304
  constexpr int ATSZ = 64 * LDK;
  constexpr int BTSZ = 192 * LDK;

  const int tid  = threadIdx.x;
  const int lane = tid & 31;
  const int wave = tid >> 5;                 // 0..3
  const int m0   = blockIdx.x * 64;

  fill_hadamard(Hbf, tid, 128);

  v8f acc[12];
  for (int nt = 0; nt < 12; ++nt)
    for (int e = 0; e < 8; ++e) acc[nt][e] = 0.0f;

  float4 ra[4], rb[12];
  auto load_regs = [&](int k0) {
    for (int j = 0; j < 4; ++j) {            // A: 64 rows x 32 -> 512 float4
      int idx = tid + (j << 7);
      int r   = idx >> 3;                    // 0..63
      int c4  = (idx & 7) << 2;
      const float* pa = x + (size_t)(m0 + r) * kD + k0 + c4;
      __builtin_prefetch(pa + 32, 0, 3);
      ra[j] = *(const float4*)pa;
    }
    for (int j = 0; j < 12; ++j) {           // B: 192 rows x 32 -> 1536 float4
      int idx = tid + (j << 7);
      int r   = idx >> 3;                    // 0..191
      int c4  = (idx & 7) << 2;
      const float* pb = (r < 128) ? (wk + (size_t)r * kD + k0 + c4)
                                  : (wp + (size_t)(r - 128) * kD + k0 + c4);
      rb[j] = *(const float4*)pb;
    }
  };
  auto store_lds = [&](int buf) {
    unsigned short* A  = As + buf * ATSZ;
    unsigned short* Bt = Bs + buf * BTSZ;
    for (int j = 0; j < 4; ++j) {
      int idx = tid + (j << 7);
      int r   = idx >> 3;
      int c4  = (idx & 7) << 2;
      unsigned short* sa = A + r * LDK + c4;
      sa[0] = f2bf(ra[j].x); sa[1] = f2bf(ra[j].y); sa[2] = f2bf(ra[j].z); sa[3] = f2bf(ra[j].w);
    }
    for (int j = 0; j < 12; ++j) {
      int idx = tid + (j << 7);
      int r   = idx >> 3;
      int c4  = (idx & 7) << 2;
      unsigned short* sb = Bt + r * LDK + c4;
      sb[0] = f2bf(rb[j].x); sb[1] = f2bf(rb[j].y); sb[2] = f2bf(rb[j].z); sb[3] = f2bf(rb[j].w);
    }
  };

  constexpr int NK = kD / 32;                // 224
  load_regs(0);
  store_lds(0);
  __syncthreads();
  for (int kt = 0; kt < NK; ++kt) {
    const int cur = kt & 1;
    const bool more = (kt + 1) < NK;
    if (more) load_regs((kt + 1) * 32);
    v16bf a = frag_a(As + cur * ATSZ + wave * 16 * LDK, LDK, lane);
    v16bf bf[12];
    for (int nt = 0; nt < 12; ++nt)
      bf[nt] = frag_b(Bs + cur * BTSZ + nt * 16 * LDK, LDK, lane);
    for (int nt = 0; nt < 12; ++nt)
      acc[nt] = WMMA_BF16(a, bf[nt], acc[nt]);
    if (more) store_lds(cur ^ 1);
    __syncthreads();
  }

  const int c0 = lane & 15;

  // ---- weights out (cols 128..191), scaled by H^-0.5 * DH^-0.5 ----
  const float wscale = 0.011048543456039804f;   // 1/(8*sqrt(128))
  for (int nt = 8; nt < 12; ++nt)
    for (int j = 0; j < 8; ++j) {
      int m = m0 + wave * 16 + j + ((lane >> 4) << 3);
      wout[(size_t)m * kH + (nt - 8) * 16 + c0] = acc[nt][j] * wscale;
    }

  // ---- layernorm (per-row over 128 cols, shfl-reduce in 16-lane halves) +
  //      RoPE in registers, bf16 to LDS ----
  float gv[8], bv[8];
  for (int nt = 0; nt < 8; ++nt) { gv[nt] = g[nt * 16 + c0]; bv[nt] = bb[nt * 16 + c0]; }
  for (int j = 0; j < 8; ++j) {
    float sum = 0.0f;
    for (int nt = 0; nt < 8; ++nt) sum += acc[nt][j];
    sum += __shfl_xor(sum, 1, 32);
    sum += __shfl_xor(sum, 2, 32);
    sum += __shfl_xor(sum, 4, 32);
    sum += __shfl_xor(sum, 8, 32);
    const float mu = sum * (1.0f / 128.0f);
    float vv = 0.0f;
    for (int nt = 0; nt < 8; ++nt) { float d = acc[nt][j] - mu; vv += d * d; }
    vv += __shfl_xor(vv, 1, 32);
    vv += __shfl_xor(vv, 2, 32);
    vv += __shfl_xor(vv, 4, 32);
    vv += __shfl_xor(vv, 8, 32);
    const float rs = rsqrtf(vv * (1.0f / 128.0f) + 1e-5f);

    float nrm[8];
    for (int nt = 0; nt < 8; ++nt)
      nrm[nt] = (acc[nt][j] - mu) * rs * gv[nt] + bv[nt];

    const int r = wave * 16 + j + ((lane >> 4) << 3);
    const int s = (m0 + r) & (kS - 1);
    float c0v = fc[s * 32 + c0],      s0v = fs[s * 32 + c0];
    float c1v = fc[s * 32 + 16 + c0], s1v = fs[s * 32 + 16 + c0];
    float n0 = nrm[0] * c0v - nrm[2] * s0v;
    float n2 = nrm[0] * s0v + nrm[2] * c0v;
    float n1 = nrm[1] * c1v - nrm[3] * s1v;
    float n3 = nrm[1] * s1v + nrm[3] * c1v;
    unsigned short* kq = kbf + r * LDB + c0;
    kq[0]   = f2bf(n0);      kq[16]  = f2bf(n1);
    kq[32]  = f2bf(n2);      kq[48]  = f2bf(n3);
    kq[64]  = f2bf(nrm[4]);  kq[80]  = f2bf(nrm[5]);
    kq[96]  = f2bf(nrm[6]);  kq[112] = f2bf(nrm[7]);
  }
  __syncthreads();

  // ---- Hadamard via WMMA + store k ----
  v8f hacc[8];
  for (int nt = 0; nt < 8; ++nt)
    for (int e = 0; e < 8; ++e) hacc[nt][e] = 0.0f;
  for (int kk = 0; kk < 4; ++kk) {
    v16bf a = frag_a(kbf + wave * 16 * LDB + kk * 32, LDB, lane);
    v16bf bf[8];
    for (int nt = 0; nt < 8; ++nt)
      bf[nt] = frag_b(Hbf + nt * 16 * LDB + kk * 32, LDB, lane);
    for (int nt = 0; nt < 8; ++nt)
      hacc[nt] = WMMA_BF16(a, bf[nt], hacc[nt]);
  }
  for (int nt = 0; nt < 8; ++nt)
    for (int j = 0; j < 8; ++j) {
      int m = m0 + wave * 16 + j + ((lane >> 4) << 3);
      kout[(size_t)m * kDH + nt * 16 + c0] = hacc[nt][j];
    }
}

__global__ void endpos_kernel(const int* __restrict__ sp, float* __restrict__ eout) {
  if (threadIdx.x == 0 && blockIdx.x == 0) eout[0] = (float)(sp[0] + kS);
}

// ---------------------------------------------------------------------------
extern "C" void kernel_launch(void* const* d_in, const int* in_sizes, int n_in,
                              void* d_out, int out_size, void* d_ws, size_t ws_size,
                              hipStream_t stream) {
  (void)in_sizes; (void)n_in; (void)out_size; (void)d_ws; (void)ws_size;
  const float* x  = (const float*)d_in[0];
  const float* qr = (const float*)d_in[1];
  const float* fc = (const float*)d_in[2];
  const float* fs = (const float*)d_in[3];
  const float* wq = (const float*)d_in[4];
  const float* wk = (const float*)d_in[5];
  const float* g  = (const float*)d_in[6];
  const float* bb = (const float*)d_in[7];
  const float* wp = (const float*)d_in[8];
  const int*   sp = (const int*)d_in[9];

  float* out  = (float*)d_out;
  float* qout = out;                       // 67108864 floats
  float* wout = out + 67108864;            //   524288 floats
  float* kout = out + 67633152;            //  1048576 floats
  float* eout = out + 68681728;            //        1

  q_fused_kernel<<<dim3(64 * 64), dim3(256), 114688, stream>>>(qr, wq, fc, fs, qout);
  kw_fused_kernel<<<dim3(128), dim3(128), 98304, stream>>>(x, wk, wp, g, bb, fc, fs, wout, kout);
  endpos_kernel<<<dim3(1), dim3(32), 0, stream>>>(sp, eout);
}
